// MultiHeadAttention_35940286333248
// MI455X (gfx1250) — matrix operand order repaired and checked by reading, not compiled, not run on hardware
//
#include <hip/hip_runtime.h>

typedef _Float16 half8  __attribute__((ext_vector_type(8)));
typedef _Float16 half16 __attribute__((ext_vector_type(16)));
typedef float    v8f    __attribute__((ext_vector_type(8)));
typedef unsigned u32x4  __attribute__((ext_vector_type(4)));
typedef unsigned u32x8  __attribute__((ext_vector_type(8)));

#define B_  2
#define N_  2048
#define D_  1024
#define H_  16
#define HC_ 64
#define R_  (B_ * N_)

// ---------------------------------------------------------------- TDM helper
// Issue a 2D tensor DMA: global (rows of td0 f16 elems, td1 rows, row stride
// stride0 elems) -> LDS at byte offset lds_off, with per-row LDS padding.
// pad_int: interval code (3 = every 64B, 4 = every 128B); pad_amt: 3 = 16B pad.
// D# layout per CDNA5 ISA ch.8 (group0 128b, group1 256b). Tracked by TENSORcnt.
__device__ __forceinline__ void tdm_load_2d(unsigned lds_off, const void* g,
                                            unsigned td0, unsigned td1,
                                            unsigned stride0,
                                            unsigned pad_int, unsigned pad_amt) {
  unsigned long long ga = (unsigned long long)(uintptr_t)g;
  u32x4 g0;
  g0[0] = 1u;                                          // count=1, user desc
  g0[1] = lds_off;                                     // lds_addr (bytes)
  g0[2] = (unsigned)ga;                                // global_addr[31:0]
  g0[3] = (unsigned)((ga >> 32) & 0x01ffffffu) | (2u << 30);  // addr[56:32], type=2
  u32x8 g1;
  g1[0] = (1u << 16)                                   // data_size = 2 bytes
        | (1u << 20) | (pad_int << 22) | (pad_amt << 25); // pad_enable
  g1[1] = (td0 & 0xffffu) << 16;                       // tensor_dim0[15:0]
  g1[2] = (td0 >> 16) | ((td1 & 0xffffu) << 16);       // tensor_dim0[31:16], dim1 lo
  g1[3] = (td1 >> 16) | (td0 << 16);                   // dim1 hi, tile_dim0
  g1[4] = td1 & 0xffffu;                               // tile_dim1 (tile_dim2=0)
  g1[5] = stride0;                                     // tensor_dim0_stride lo32
  g1[6] = 0u;                                          // stride hi, dim1_stride lo
  g1[7] = 0u;
  asm volatile("tensor_load_to_lds %0, %1" :: "s"(g0), "s"(g1) : "memory");
}

__device__ __forceinline__ unsigned lds_off_of(const void* p) {
  return (unsigned)(uintptr_t)p;    // LDS aperture: addr[31:0] = LDS byte offset
}

// ---------------------------------------------------------------- converts
__global__ void cvt_f16(const float* __restrict__ src, _Float16* __restrict__ dst, int n) {
  for (int i = (blockIdx.x * blockDim.x + threadIdx.x) * 8; i < n;
       i += gridDim.x * blockDim.x * 8) {
    const float4 a = *(const float4*)(src + i);
    const float4 b = *(const float4*)(src + i + 4);
    half8 o = { (_Float16)a.x, (_Float16)a.y, (_Float16)a.z, (_Float16)a.w,
                (_Float16)b.x, (_Float16)b.y, (_Float16)b.z, (_Float16)b.w };
    *(half8*)(dst + i) = o;
  }
}

// dst[n][k] = src[k][n]  (DxD weight, stored column-major so GEMM B reads rows)
__global__ void cvt_f16_T(const float* __restrict__ src, _Float16* __restrict__ dst) {
  for (int i = (blockIdx.x * blockDim.x + threadIdx.x) * 8; i < D_ * D_;
       i += gridDim.x * blockDim.x * 8) {
    const int nn = i >> 10, k0 = i & (D_ - 1);
    half8 o;
#pragma unroll
    for (int e = 0; e < 8; ++e) o[e] = (_Float16)src[(size_t)(k0 + e) * D_ + nn];
    *(half8*)(dst + i) = o;
  }
}

// ---------------------------------------------------------------- GEMM 64x64 tile
// A: [R_, D_] f16 row-major. Bt: [D_, D_] f16, Bt[n][k] = W[k][n].
// MODE 0: V (plain -> head buffer), 1: K (rope), 2: Q (scale+rope), 3: out-proj (f32)
// Tiles staged by the Tensor Data Mover, double-buffered (DMA of chunk k+1
// overlaps WMMA on chunk k); LDS pitch 40 halfs = 80B (conflict-free b128).
template <int MODE>
__global__ __launch_bounds__(128) void gemm64(const _Float16* __restrict__ A,
                                              const _Float16* __restrict__ Bt,
                                              const float* __restrict__ bias,
                                              _Float16* __restrict__ outH,
                                              float* __restrict__ outF) {
  const int rb   = blockIdx.x * 64;
  const int cb   = blockIdx.y * 64;
  const int tid  = threadIdx.x;
  const int wave = tid >> 5;
  const int lane = tid & 31;
  const int lrow = lane & 15;
  const int hs   = lane >> 4;          // 0 = lanes 0-15, 1 = lanes 16-31

  __shared__ __align__(16) _Float16 As[2][64][40];
  __shared__ __align__(16) _Float16 Bs[2][64][40];

  v8f acc[4] = {};

  // prologue: DMA first k-chunk (rows 64 x 32 halfs, pad 64B->+16B => 80B pitch)
  if (wave == 0) {
    tdm_load_2d(lds_off_of(&As[0][0][0]), A  + (size_t)rb * D_, 32, 64, D_, 3, 3);
    tdm_load_2d(lds_off_of(&Bs[0][0][0]), Bt + (size_t)cb * D_, 32, 64, D_, 3, 3);
    __builtin_amdgcn_s_wait_tensorcnt(0);
  }
  __syncthreads();

  for (int k0 = 0; k0 < D_; k0 += 32) {
    const int cur = (k0 >> 5) & 1;
    if (wave == 0 && (k0 + 32) < D_) {          // async DMA of next chunk
      tdm_load_2d(lds_off_of(&As[cur ^ 1][0][0]), A  + (size_t)rb * D_ + k0 + 32, 32, 64, D_, 3, 3);
      tdm_load_2d(lds_off_of(&Bs[cur ^ 1][0][0]), Bt + (size_t)cb * D_ + k0 + 32, 32, 64, D_, 3, 3);
    }

    // A fragment 16x32: lane<16 -> K 0-7,16-23 ; lane>=16 -> K 8-15,24-31
    half16 af;
    *(half8*)&af       = *(const half8*)&As[cur][wave * 16 + lrow][hs * 8];
    *((half8*)&af + 1) = *(const half8*)&As[cur][wave * 16 + lrow][hs * 8 + 16];
#pragma unroll
    for (int t = 0; t < 4; ++t) {
      // B fragment 32x16: lane holds column (= Bt row), lane<16 -> K 0-7,16-23 ; lane>=16 -> K 8-15,24-31
      half16 bf;
      *(half8*)&bf       = *(const half8*)&Bs[cur][t * 16 + lrow][hs * 8];
      *((half8*)&bf + 1) = *(const half8*)&Bs[cur][t * 16 + lrow][hs * 8 + 16];
      acc[t] = __builtin_amdgcn_wmma_f32_16x16x32_f16(false, af, false, bf,
                                                      (short)0, acc[t], false, false);
    }

    if (wave == 0) __builtin_amdgcn_s_wait_tensorcnt(0);   // next chunk landed
    __syncthreads();
  }

  // epilogue: C layout -> lane col = cb + t*16 + lrow ; row = rb + wave*16 + hs*8 + r
#pragma unroll
  for (int t = 0; t < 4; ++t) {
    const int col = cb + t * 16 + lrow;
    const float bval = bias[col];
#pragma unroll
    for (int r = 0; r < 8; ++r) {
      const int row = rb + wave * 16 + hs * 8 + r;
      float v = acc[t][r] + bval;
      if constexpr (MODE == 3) {
        outF[(size_t)row * D_ + col] = v;
      } else {
        if constexpr (MODE == 2) v *= 0.125f;            // HC^-0.5
        if constexpr (MODE == 1 || MODE == 2) {
          const float partner = __shfl_xor(v, 1, 32);    // adjacent column = pair partner
          const int c = col & (HC_ - 1);
          const int p = c >> 1;
          const float freq = __expf(-(float)(2 * p) * (9.210340371976184f / 64.0f));
          const float ang  = (float)(row & (N_ - 1)) * freq;
          float sn, cs;
          sincosf(ang, &sn, &cs);
          v = (c & 1) ? (v * cs + partner * sn) : (v * cs - partner * sn);
        }
        const int bb  = row >> 11;          // / N_
        const int pos = row & (N_ - 1);
        const int hh  = col >> 6;
        const int cc  = col & (HC_ - 1);
        outH[(((size_t)bb * H_ + hh) * N_ + pos) * HC_ + cc] = (_Float16)v;
      }
    }
  }
}

// ---------------------------------------------------------------- flash attention
// Q,K,V: [B*H][N][64] f16.  O: [B*N][D] f16 (heads merged).
// Q tile + double-buffered K tiles arrive via TDM (row pad 128B->+16B => 144B
// pitch); V is transposed through VGPRs into Vt for the V^T * P^T WMMA.
__global__ __launch_bounds__(128) void attn_kernel(const _Float16* __restrict__ Q,
                                                   const _Float16* __restrict__ K,
                                                   const _Float16* __restrict__ V,
                                                   const unsigned char* __restrict__ pad,
                                                   _Float16* __restrict__ O) {
  const int bh   = blockIdx.x;
  const int b    = bh >> 4;
  const int h    = bh & (H_ - 1);
  const int qb   = blockIdx.y * 64;
  const int tid  = threadIdx.x;
  const int wave = tid >> 5;
  const int lane = tid & 31;
  const int lrow = lane & 15;
  const int hs   = lane >> 4;

  __shared__ __align__(16) _Float16 Qs[64][72];       // 144B pitch (16B aligned)
  __shared__ __align__(16) _Float16 Ks[2][64][72];    // double-buffered K tiles
  __shared__ __align__(16) _Float16 Vt[64][72];       // [channel][key]
  __shared__ __align__(16) _Float16 Pw[4][16][72];    // per-wave P [query][key]
  __shared__ __align__(8)  unsigned char msk[64];

  const size_t base = (size_t)bh * N_;
  const int ldr = tid >> 1;
  const int ldc = (tid & 1) * 32;

  // prologue: DMA Q tile and first K tile
  if (wave == 0) {
    tdm_load_2d(lds_off_of(&Qs[0][0]),    Q + (base + qb) * HC_, HC_, 64, HC_, 4, 3);
    tdm_load_2d(lds_off_of(&Ks[0][0][0]), K + base * HC_,        HC_, 64, HC_, 4, 3);
    __builtin_amdgcn_s_wait_tensorcnt(0);
  }
  __syncthreads();

  // preload Q as B-fragments of S^T = K * Q^T  (lane = query column, table K order)
  half16 qf[2];
#pragma unroll
  for (int kk = 0; kk < 2; ++kk) {
    *(half8*)&qf[kk]       = *(const half8*)&Qs[wave * 16 + lrow][kk * 32 + hs * 8];
    *((half8*)&qf[kk] + 1) = *(const half8*)&Qs[wave * 16 + lrow][kk * 32 + hs * 8 + 16];
  }

  float m = -3.0e38f, l = 0.0f;
  v8f oa[4] = {};
  const int qg = qb + wave * 16 + lrow;   // this lane's query (both lane halves)

  for (int jb = 0; jb <= (int)blockIdx.y; ++jb) {
    const int kb  = jb * 64;
    const int cur = jb & 1;

    // async DMA of next K tile into the other buffer (overlaps this iteration)
    if (wave == 0 && jb < (int)blockIdx.y)
      tdm_load_2d(lds_off_of(&Ks[cur ^ 1][0][0]),
                  K + (base + kb + 64) * HC_, HC_, 64, HC_, 4, 3);

    {   // V tile -> transposed Vt, plus pad mask
      const _Float16* Vg = V + (base + kb + ldr) * HC_ + ldc;
#pragma unroll
      for (int i = 0; i < 32; i += 8) {
        half8 vv = *(const half8*)(Vg + i);
#pragma unroll
        for (int e = 0; e < 8; ++e) Vt[ldc + i + e][ldr] = vv[e];
      }
      if (tid < 64) msk[tid] = pad[(size_t)b * N_ + kb + tid];
    }
    __syncthreads();

    if (jb < (int)blockIdx.y)             // warm caches for next V tile
      __builtin_prefetch(V + (base + kb + 64) * HC_ + tid * 32, 0, 1);

    // pad-mask bytes for this lane's key rows: 4 x u64 (branchless masking)
    unsigned long long mb[4];
#pragma unroll
    for (int t = 0; t < 4; ++t)
      mb[t] = *(const unsigned long long*)&msk[t * 16 + hs * 8];

    // S^T tiles: rows = keys (4 subtiles of 16), cols = 16 queries of this wave
    v8f s[4];
#pragma unroll
    for (int t = 0; t < 4; ++t) {
      v8f c = {};
#pragma unroll
      for (int kk = 0; kk < 2; ++kk) {
        half16 af;   // K rows t*16+lrow, head-channel chunk kk*32
        *(half8*)&af       = *(const half8*)&Ks[cur][t * 16 + lrow][kk * 32 + hs * 8];
        *((half8*)&af + 1) = *(const half8*)&Ks[cur][t * 16 + lrow][kk * 32 + hs * 8 + 16];
        c = __builtin_amdgcn_wmma_f32_16x16x32_f16(false, af, false, qf[kk],
                                                   (short)0, c, false, false);
      }
      s[t] = c;
    }

    // causal + pad mask (branchless), block max, fold the 16-lane halves
    float bm = -3.0e38f;
#pragma unroll
    for (int t = 0; t < 4; ++t)
#pragma unroll
      for (int r = 0; r < 8; ++r) {
        const int kl = t * 16 + hs * 8 + r;
        const int dead = (int)((kb + kl) > qg) | (int)((mb[t] >> (8 * r)) & 1u);
        const float sv = dead ? -3.0e38f : s[t][r];
        s[t][r] = sv;
        bm = fmaxf(bm, sv);
      }
    bm = fmaxf(bm, __shfl_xor(bm, 16, 32));
    const float mn   = fmaxf(m, bm);
    const float resc = __expf(m - mn);
    m = mn;

    float ps = 0.0f;
#pragma unroll
    for (int t = 0; t < 4; ++t) {
      half8 pv;                                  // 8 consecutive keys -> one b128
#pragma unroll
      for (int r = 0; r < 8; ++r) {
        const float p = __expf(s[t][r] - mn);
        ps += p;
        pv[r] = (_Float16)p;
      }
      *(half8*)&Pw[wave][lrow][t * 16 + hs * 8] = pv;
    }
    ps += __shfl_xor(ps, 16, 32);
    l = l * resc + ps;
#pragma unroll
    for (int t = 0; t < 4; ++t)
#pragma unroll
      for (int r = 0; r < 8; ++r) oa[t][r] *= resc;

    // O^T += V^T * P^T   (rows = channels, cols = queries)
#pragma unroll
    for (int t = 0; t < 4; ++t) {
#pragma unroll
      for (int kk = 0; kk < 2; ++kk) {
        half16 af, bf;
        *(half8*)&af       = *(const half8*)&Vt[t * 16 + lrow][kk * 32 + hs * 8];
        *((half8*)&af + 1) = *(const half8*)&Vt[t * 16 + lrow][kk * 32 + hs * 8 + 16];
        *(half8*)&bf       = *(const half8*)&Pw[wave][lrow][kk * 32 + hs * 8];
        *((half8*)&bf + 1) = *(const half8*)&Pw[wave][lrow][kk * 32 + hs * 8 + 16];
        oa[t] = __builtin_amdgcn_wmma_f32_16x16x32_f16(false, af, false, bf,
                                                       (short)0, oa[t], false, false);
      }
    }

    if (wave == 0) __builtin_amdgcn_s_wait_tensorcnt(0);   // next K tile landed
    __syncthreads();
  }

  const float inv = 1.0f / l;
  _Float16* Op = O + ((size_t)b * N_ + qg) * D_ + h * HC_;
#pragma unroll
  for (int t = 0; t < 4; ++t) {
    half8 ov;                                    // 8 consecutive channels -> one b128
#pragma unroll
    for (int r = 0; r < 8; ++r) ov[r] = (_Float16)(oa[t][r] * inv);
    *(half8*)(Op + t * 16 + hs * 8) = ov;
  }
}

// ---------------------------------------------------------------- launch
extern "C" void kernel_launch(void* const* d_in, const int* in_sizes, int n_in,
                              void* d_out, int out_size, void* d_ws, size_t ws_size,
                              hipStream_t stream) {
  const float* x_q  = (const float*)d_in[0];
  const float* x_kv = (const float*)d_in[1];
  const unsigned char* pad = (const unsigned char*)d_in[2];
  const float* Wq = (const float*)d_in[3];
  const float* bq = (const float*)d_in[4];
  const float* Wk = (const float*)d_in[5];
  const float* bk = (const float*)d_in[6];
  const float* Wv = (const float*)d_in[7];
  const float* bv = (const float*)d_in[8];
  const float* Wo = (const float*)d_in[9];
  const float* bo = (const float*)d_in[10];
  float* out = (float*)d_out;

  const size_t XE = (size_t)R_ * D_;   // 4M elems
  const size_t WE = (size_t)D_ * D_;   // 1M elems
  _Float16* xq16  = (_Float16*)d_ws;
  _Float16* xkv16 = xq16  + XE;
  _Float16* WqT   = xkv16 + XE;
  _Float16* WkT   = WqT + WE;
  _Float16* WvT   = WkT + WE;
  _Float16* WoT   = WvT + WE;
  _Float16* Q16   = WoT + WE;
  _Float16* K16   = Q16 + XE;
  _Float16* V16   = K16 + XE;
  _Float16* O16   = V16 + XE;

  cvt_f16<<<1024, 256, 0, stream>>>(x_q,  xq16,  (int)XE);
  cvt_f16<<<1024, 256, 0, stream>>>(x_kv, xkv16, (int)XE);
  cvt_f16_T<<<512, 256, 0, stream>>>(Wq, WqT);
  cvt_f16_T<<<512, 256, 0, stream>>>(Wk, WkT);
  cvt_f16_T<<<512, 256, 0, stream>>>(Wv, WvT);
  cvt_f16_T<<<512, 256, 0, stream>>>(Wo, WoT);

  dim3 g(R_ / 64, D_ / 64);
  gemm64<2><<<g, 128, 0, stream>>>(xq16,  WqT, bq, Q16, nullptr);  // Q: scale+rope
  gemm64<1><<<g, 128, 0, stream>>>(xkv16, WkT, bk, K16, nullptr);  // K: rope
  gemm64<0><<<g, 128, 0, stream>>>(xkv16, WvT, bv, V16, nullptr);  // V

  attn_kernel<<<dim3(B_ * H_, N_ / 64), 128, 0, stream>>>(Q16, K16, V16, pad, O16);

  gemm64<3><<<g, 128, 0, stream>>>(O16, WoT, bo, nullptr, out);    // out-proj -> f32
}
